// LSTM_4080218931289
// MI455X (gfx1250) — matrix operand order repaired and checked by reading, not compiled, not run on hardware
//
#include <hip/hip_runtime.h>
#include <hip/hip_bf16.h>

// Problem sizes
#define T_STEPS 2048
#define BATCH   32
#define ISZ     512
#define HSZ     512
#define NWG     32                  // persistent workgroups (one per WGP)
#define HPW     (HSZ / NWG)         // 16 hidden units per WG
#define GROWS   (4 * HPW)           // 64 gate rows per WG (i,f,o,g x 16)

// Workspace layout (byte offsets into d_ws)
#define OFF_CNT 0
#define OFF_H0  4096                            // h broadcast buf 0: 32x512 bf16 = 32768 B
#define OFF_H1  (OFF_H0 + 32768)                // h broadcast buf 1
#define OFF_W   73728                           // packed weights: 2048x1024 bf16 = 4 MB
#define OFF_X   (OFF_W + 4 * 1024 * 1024)       // x in bf16: 64 MB
// total ws needed ~ 68.1 MB

// LDS layout (byte offsets into dynamic shared)
#define L_W     0                               // 128 KB packed weight fragments
#define L_X0    131072                          // 32 KB x tile, slot 0 (32 x 1024 B)
#define L_X1    163840                          // 32 KB x tile, slot 1
#define L_H     196608                          // 32 KB h tile (32 x 1024 B)
#define L_PRE   229376                          // 8 KB: 32 x 64 f32 preact
#define L_BIAS  237568                          // 256 B
#define L_C     237824                          // 2 KB: 32 x 16 f32 cell state
#define SMEM_BYTES 239872

typedef __attribute__((ext_vector_type(16))) __bf16 v16bf;
typedef __attribute__((ext_vector_type(8)))  float  v8f;
typedef __attribute__((ext_vector_type(4)))  unsigned v4u;
typedef __attribute__((ext_vector_type(8)))  int      v8i;
typedef __attribute__((ext_vector_type(4)))  int      v4i;

union FragU { uint4 u[2]; v16bf v; };

#if defined(__has_builtin)
#if __has_builtin(__builtin_amdgcn_tensor_load_to_lds)
#define USE_TDM 1
#endif
#endif
#ifndef USE_TDM
#define USE_TDM 0
#endif

#if USE_TDM
// 1-D contiguous global -> LDS copy via the Tensor Data Mover.
// nbytes must be a multiple of 8; counted as 8-byte elements so tile_dim0
// (16 bits) covers up to 512 KB. D# fields per CDNA5 ISA ch.8.
// This toolchain exposes the 6-arg builtin:
//   (uint32x4 g0, int32x8 g1, int32x4 g2, int32x4 g3, int32x8 g4, i32 cpol)
__device__ __forceinline__ void tdm_copy_1d(unsigned lds_off, const void* gsrc,
                                            unsigned nbytes) {
    unsigned long long ga = (unsigned long long)(size_t)gsrc;
    unsigned n8 = nbytes >> 3;
    v4u g0;
    g0[0] = 1u;                                           // count=1 (user mode)
    g0[1] = lds_off;                                      // lds_addr
    g0[2] = (unsigned)(ga & 0xFFFFFFFFu);                 // global_addr[31:0]
    g0[3] = (unsigned)((ga >> 32) & 0x01FFFFFFu)          // global_addr[56:32]
            | (2u << 30);                                 // type = 2 (image)
    v8i g1;
    g1[0] = (int)(3u << 16);                              // data_size=3 (8B), mask=0
    g1[1] = (int)((n8 & 0xFFFFu) << 16);                  // tensor_dim0[15:0]
    g1[2] = (int)((n8 >> 16) | (1u << 16));               // tensor_dim0[31:16], tensor_dim1=1
    g1[3] = (int)(n8 << 16);                              // tile_dim0 = n8
    g1[4] = 1;                                            // tile_dim1 = 1, tile_dim2 = 0
    g1[5] = (int)n8;                                      // tensor_dim0_stride[31:0]
    g1[6] = 0;
    g1[7] = 0;
    v4i z4 = {0, 0, 0, 0};                                // groups 2/3 unused (<=2D)
    v8i z8 = {0, 0, 0, 0, 0, 0, 0, 0};
    __builtin_amdgcn_tensor_load_to_lds(g0, g1, z4, z4, z8, 0);
}
#endif

__device__ __forceinline__ float fast_tanh(float x) {
#if defined(__has_builtin) && __has_builtin(__builtin_amdgcn_tanhf)
    return __builtin_amdgcn_tanhf(x);
#elif defined(__has_builtin) && __has_builtin(__builtin_amdgcn_tanh_f32)
    return __builtin_amdgcn_tanh_f32(x);
#else
    return tanhf(x);
#endif
}

__device__ __forceinline__ float fast_sigmoid(float x) {
    return 1.f / (1.f + __expf(-x));
}

// ---------------------------------------------------------------------------
// Setup: zero barrier counter and both h broadcast buffers (h0 = 0).
__global__ void lstm_init_ws(unsigned* cnt, uint4* wsh01 /* 2 x 2048 uint4 */) {
    if (threadIdx.x == 0) *cnt = 0u;
    uint4 z; z.x = z.y = z.z = z.w = 0u;
    for (int i = threadIdx.x; i < 4096; i += blockDim.x) wsh01[i] = z;
}

// ---------------------------------------------------------------------------
// Pack [Wi | Wh] (fp32 row-major, rows = 4H gate rows, K = 1024) into bf16
// WMMA B-fragment layout:
//   fragment id f = g*128 + nt*32 + ks   (g = owning WG, nt = gate, ks = k-step)
//   within a fragment: lane L owns 32 contiguous bytes = 16 bf16:
//     column n = nt_rowbase + (L&15), K run = ks*32 + (L>>4)*16 .. +16
__global__ void lstm_pack_weights(const float* __restrict__ Wi,
                                  const float* __restrict__ Wh,
                                  __hip_bfloat16* __restrict__ wsw) {
    int id   = blockIdx.x * blockDim.x + threadIdx.x;   // 0 .. 131071
    int g    = id >> 12;
    int rem  = id & 4095;
    int nt   = rem >> 10;
    int rem2 = rem & 1023;
    int ks   = rem2 >> 5;
    int lane = rem2 & 31;
    int row  = nt * HSZ + g * HPW + (lane & 15);        // gate row in 0..2047
    int k0   = ks * 32 + (lane >> 4) * 16;

    alignas(16) __hip_bfloat16 tmp[16];
#pragma unroll
    for (int k = 0; k < 16; ++k) {
        int kk = k0 + k;
        float v = (kk < ISZ) ? Wi[row * ISZ + kk] : Wh[row * HSZ + (kk - ISZ)];
        tmp[k] = __float2bfloat16(v);
    }
    uint4* dst = (uint4*)(wsw + (size_t)id * 16);
    dst[0] = ((uint4*)tmp)[0];
    dst[1] = ((uint4*)tmp)[1];
}

// ---------------------------------------------------------------------------
// Convert x (T,B,I) fp32 -> bf16, same layout. 8 elems / thread.
__global__ void lstm_convert_x(const float* __restrict__ x,
                               __hip_bfloat16* __restrict__ wsx) {
    size_t i0 = (size_t)(blockIdx.x * blockDim.x + threadIdx.x) * 8;
    const float4* s = (const float4*)(x + i0);
    float4 a = s[0], b = s[1];
    alignas(16) __hip_bfloat16 t[8] = {
        __float2bfloat16(a.x), __float2bfloat16(a.y),
        __float2bfloat16(a.z), __float2bfloat16(a.w),
        __float2bfloat16(b.x), __float2bfloat16(b.y),
        __float2bfloat16(b.z), __float2bfloat16(b.w)};
    *(uint4*)(wsx + i0) = *(uint4*)t;
}

// ---------------------------------------------------------------------------
// Persistent LSTM kernel. 32 WGs x 256 threads (8 wave32s). Each WG owns
// 16 hidden units; per step does a 32x1024 @ 1024x64 bf16 WMMA GEMM with
// weights resident in LDS. TDM prefetches x_{t+1} during the GEMM and pulls
// the h broadcast right after the device-wide barrier.
__global__ __launch_bounds__(256) void lstm_persistent(
    const __hip_bfloat16* __restrict__ wsx,     // (T,B,I) bf16
    __hip_bfloat16* __restrict__ wsh,           // 2 x (B,H) bf16 double buffer
    const __hip_bfloat16* __restrict__ wsw,     // packed weights
    unsigned* cnt,
    const float* __restrict__ bi,
    const float* __restrict__ bh,
    float* __restrict__ out)
{
    extern __shared__ char smem[];
    float* spre  = (float*)(smem + L_PRE);
    float* sbias = (float*)(smem + L_BIAS);
    float* sc    = (float*)(smem + L_C);

    const int tid   = threadIdx.x;
    const int g     = blockIdx.x;
    const int lane  = tid & 31;
    const int wv    = tid >> 5;     // wave 0..7
    const int mtile = wv >> 2;      // 0..1  (batch tile)
    const int nt    = wv & 3;       // 0..3  (gate)

#if USE_TDM
    const unsigned ldsbase = (unsigned)(size_t)(void*)smem;
    if (wv == 0) {
        // weights (128 KB), x_0 tile, h_0 tile — all on the TDM engine
        tdm_copy_1d(ldsbase + L_W, (const char*)wsw + (size_t)g * 131072u, 131072u);
        tdm_copy_1d(ldsbase + L_X0, (const char*)wsx, 32768u);
        tdm_copy_1d(ldsbase + L_H, (const char*)wsh, 32768u);
        __builtin_amdgcn_s_wait_tensorcnt(0);
    }
#else
    {
        const uint4* src = (const uint4*)wsw + (size_t)g * 8192;
        uint4* dst = (uint4*)(smem + L_W);
        for (int i = tid; i < 8192; i += 256) dst[i] = src[i];
    }
#endif
    // Combined bias bi+bh for our 64 gate rows.
    if (tid < GROWS) {
        int ntl = tid >> 4, col = tid & 15;
        int row = ntl * HSZ + g * HPW + col;
        sbias[tid] = bi[row] + bh[row];
    }
    // c0 = 0
    for (int i = tid; i < BATCH * HPW; i += 256) sc[i] = 0.f;
    __syncthreads();

    const int arow  = mtile * 16 + (lane & 15);
    const int aoff0 = (lane >> 4) * 32;       // A-fragment K sub-run select
    float* outh = out + (size_t)T_STEPS * BATCH * HSZ;
    float* outc = outh + BATCH * HSZ;

    for (int t = 0; t < T_STEPS; ++t) {
#if USE_TDM
        // Prefetch x_{t+1} into the other LDS slot; overlaps the WMMAs below.
        if (wv == 0 && t + 1 < T_STEPS) {
            tdm_copy_1d(ldsbase + (((t + 1) & 1) ? L_X1 : L_X0),
                        (const char*)wsx + (size_t)(t + 1) * 32768u, 32768u);
        }
#else
        {   // manual staging fallback: x_t -> slot t&1, h_t -> L_H
            const uint4* sx = (const uint4*)((const char*)wsx + (size_t)t * 32768u);
            const uint4* sh = (const uint4*)((const char*)wsh + (size_t)(t & 1) * 32768u);
            uint4* dx = (uint4*)(smem + ((t & 1) ? L_X1 : L_X0));
            uint4* dh = (uint4*)(smem + L_H);
#pragma unroll
            for (int j = 0; j < 8; ++j) {
                int i = tid + j * 256;
                dx[i] = sx[i];
                dh[i] = sh[i];
            }
        }
        __syncthreads();
#endif

        // ---- WMMA GEMM: this wave computes one 16x16 f32 tile ----
        v8f acc = {};
        {
            const char* bbase = smem + L_W + (size_t)nt * 32768 + lane * 32;
            const char* axb = smem + ((t & 1) ? L_X1 : L_X0) + arow * 1024 + aoff0;
            const char* ahb = smem + L_H + arow * 1024 + aoff0;
#pragma unroll 4
            for (int ks = 0; ks < 16; ++ks) {       // K = 0..511  (x part)
                FragU fa, fb;
                fa.u[0] = *(const uint4*)(axb + ks * 64);
                fa.u[1] = *(const uint4*)(axb + ks * 64 + 16);
                fb.u[0] = *(const uint4*)(bbase + ks * 1024);
                fb.u[1] = *(const uint4*)(bbase + ks * 1024 + 16);
                acc = __builtin_amdgcn_wmma_f32_16x16x32_bf16(
                    false, fa.v, false, fb.v, (short)0, acc, false, false);
            }
#pragma unroll 4
            for (int ks = 0; ks < 16; ++ks) {       // K = 512..1023 (h part)
                FragU fa, fb;
                fa.u[0] = *(const uint4*)(ahb + ks * 64);
                fa.u[1] = *(const uint4*)(ahb + ks * 64 + 16);
                fb.u[0] = *(const uint4*)(bbase + (16 + ks) * 1024);
                fb.u[1] = *(const uint4*)(bbase + (16 + ks) * 1024 + 16);
                acc = __builtin_amdgcn_wmma_f32_16x16x32_bf16(
                    false, fa.v, false, fb.v, (short)0, acc, false, false);
            }
        }
        // Scatter D tile (16x16 f32) to preact: lanes 0-15 hold M=v, 16-31 M=8+v.
        {
            int ncol  = lane & 15;
            int mbase = mtile * 16 + ((lane >> 4) << 3);
#pragma unroll
            for (int v = 0; v < 8; ++v)
                spre[(mbase + v) * GROWS + nt * 16 + ncol] = acc[v];
        }
        __syncthreads();

        // ---- gate math: 512 (batch, hidden) pairs, 2 per thread ----
        __hip_bfloat16* hnext = wsh + (size_t)((t + 1) & 1) * 16384;
#pragma unroll
        for (int rr = 0; rr < 2; ++rr) {
            int r = tid + rr * 256;
            int b = r >> 4, col = r & 15;
            const float* pb = spre + b * GROWS;
            float iv = fast_sigmoid(pb[col]      + sbias[col]);
            float fv = fast_sigmoid(pb[16 + col] + sbias[16 + col]);
            float ov = fast_sigmoid(pb[32 + col] + sbias[32 + col]);
            float gv = fast_tanh(pb[48 + col] + sbias[48 + col]);
            float c_new = sc[r] * fv + iv * gv;
            float h_new = ov * fast_tanh(c_new);
            sc[r] = c_new;
            int hid = g * HPW + col;
            out[((size_t)t * BATCH + b) * HSZ + hid] = h_new;
            hnext[b * HSZ + hid] = __float2bfloat16(h_new);
            if (t == T_STEPS - 1) {
                outh[b * HSZ + hid] = h_new;
                outc[b * HSZ + hid] = c_new;
            }
        }

        // ---- device-wide barrier (monotonic counter, release/acquire) ----
        __syncthreads();
        if (tid == 0) {
            __hip_atomic_fetch_add(cnt, 1u, __ATOMIC_RELEASE,
                                   __HIP_MEMORY_SCOPE_AGENT);
            unsigned target = (unsigned)(t + 1) * NWG;
            while (__hip_atomic_load(cnt, __ATOMIC_ACQUIRE,
                                     __HIP_MEMORY_SCOPE_AGENT) < target) {
                __builtin_amdgcn_s_sleep(8);
            }
        }
        __syncthreads();

#if USE_TDM
        // Pull the fresh h broadcast (written by all WGs before the barrier).
        if (wv == 0 && t + 1 < T_STEPS) {
            tdm_copy_1d(ldsbase + L_H,
                        (const char*)wsh + (size_t)((t + 1) & 1) * 32768u, 32768u);
            __builtin_amdgcn_s_wait_tensorcnt(0);
        }
        __syncthreads();
#endif
    }
}

// ---------------------------------------------------------------------------
extern "C" void kernel_launch(void* const* d_in, const int* in_sizes, int n_in,
                              void* d_out, int out_size, void* d_ws, size_t ws_size,
                              hipStream_t stream) {
    const float* x  = (const float*)d_in[0];
    const float* Wi = (const float*)d_in[1];
    const float* bi = (const float*)d_in[2];
    const float* Wh = (const float*)d_in[3];
    const float* bh = (const float*)d_in[4];
    float* out = (float*)d_out;
    char*  ws  = (char*)d_ws;

    unsigned*       cnt = (unsigned*)(ws + OFF_CNT);
    __hip_bfloat16* wsh = (__hip_bfloat16*)(ws + OFF_H0);   // double buffer
    __hip_bfloat16* wsw = (__hip_bfloat16*)(ws + OFF_W);
    __hip_bfloat16* wsx = (__hip_bfloat16*)(ws + OFF_X);

    (void)in_sizes; (void)n_in; (void)out_size; (void)ws_size;

    (void)hipFuncSetAttribute(reinterpret_cast<const void*>(lstm_persistent),
                              hipFuncAttributeMaxDynamicSharedMemorySize,
                              SMEM_BYTES);

    lstm_init_ws<<<1, 256, 0, stream>>>(cnt, (uint4*)wsh);
    lstm_pack_weights<<<512, 256, 0, stream>>>(Wi, Wh, wsw);
    // T*B*I / (256 threads * 8 elems) = 16384 blocks
    lstm_convert_x<<<16384, 256, 0, stream>>>(x, wsx);
    lstm_persistent<<<NWG, 256, SMEM_BYTES, stream>>>(wsx, wsh, wsw, cnt,
                                                      bi, bh, out);
}
